// Conv3dNet_15633680958103
// MI455X (gfx1250) — matrix operand order repaired and checked by reading, compiled, not run on hardware
//
#include <hip/hip_runtime.h>
#include <math.h>

typedef _Float16 half16 __attribute__((ext_vector_type(16)));
typedef float    float8 __attribute__((ext_vector_type(8)));

#define GPTS 5832   // 18^3
#define NATOMS_PB 512
#define NB_BATCH 64

// ---------------------------------------------------------------------------
// Gaussian blur / segment-sum: one block per batch.
// Atom slab (512 x 5 floats = 10KB) staged into LDS with CDNA5 async
// global->LDS B128 copies (ASYNCcnt), then per-type field accumulation in LDS.
// ---------------------------------------------------------------------------
__global__ void blur_kernel(const float* __restrict__ x, float* __restrict__ fields) {
    __shared__ float xs[NATOMS_PB * 5];   // raw atom rows
    __shared__ float fld[GPTS];
    __shared__ float red[256];
    const int b = blockIdx.x, tid = threadIdx.x;

    // async-copy 10240 bytes = 640 x B128 transfers, global -> LDS
    {
        const uint64_t gbase = (uint64_t)(const char*)(x + (size_t)b * NATOMS_PB * 5);
        const uint32_t lbase = (uint32_t)(uintptr_t)&xs[0];
#pragma unroll
        for (int j = 0; j < 3; ++j) {
            int idx = tid + j * 256;
            if (idx < 640) {
                uint64_t ga = gbase + (uint64_t)idx * 16;
                uint32_t la = lbase + (uint32_t)idx * 16;
                asm volatile("global_load_async_to_lds_b128 %0, %1, off"
                             :: "v"(la), "v"(ga) : "memory");
            }
        }
        asm volatile("s_wait_asynccnt 0x0" ::: "memory");
    }
    __syncthreads();

    for (int t = 0; t < 6; ++t) {
        for (int p = tid; p < GPTS; p += 256) fld[p] = 0.0f;
        __syncthreads();
        for (int a = 0; a < NATOMS_PB; ++a) {
            if ((int)xs[a * 5 + 1] != t) continue;   // uniform branch (LDS data)
            const float px = xs[a * 5 + 2];
            const float py = xs[a * 5 + 3];
            const float pz = xs[a * 5 + 4];
            float dv[23];
            float part = 0.0f;
            int c = 0;
            for (int p = tid; p < GPTS; p += 256, ++c) {
                int ix = p / 324; int rem = p - ix * 324;
                int iy = rem / 18; int iz = rem - iy * 18;
                float dx = px - (-8.5f + (float)ix);
                float dy = py - (-8.5f + (float)iy);
                float dz = pz - (-8.5f + (float)iz);
                float d2 = dx * dx + dy * dy + dz * dz;
                float e = expf(-d2 * (1.0f / 0.72f));   // SIGP2 = 0.72
                dv[c] = e; part += e;
            }
            red[tid] = part; __syncthreads();
            for (int s = 128; s > 0; s >>= 1) {
                if (tid < s) red[tid] += red[tid + s];
                __syncthreads();
            }
            const float inv = 1.0f / (red[0] + 1e-6f);
            __syncthreads();
            c = 0;
            for (int p = tid; p < GPTS; p += 256, ++c) fld[p] += dv[c] * inv;
        }
        __syncthreads();
        float* outp = fields + (size_t)(b * 6 + t) * GPTS;
        for (int p = tid; p < GPTS; p += 256) outp[p] = fld[p];
        __syncthreads();
    }
}

// ---------------------------------------------------------------------------
// fp32 -> f16 padded weight prep: wf[Np][Kp], zero pad beyond (Nout, Kin).
// ---------------------------------------------------------------------------
__global__ void prep_w_kernel(const float* __restrict__ w, _Float16* __restrict__ wf,
                              int Nout, int Kin, int Np, int Kp) {
    int i = blockIdx.x * blockDim.x + threadIdx.x;
    int total = Np * Kp;
    if (i >= total) return;
    int n = i / Kp; int k = i - n * Kp;
    float v = (n < Nout && k < Kin) ? w[(size_t)n * Kin + k] : 0.0f;
    wf[i] = (_Float16)v;
}

// ---------------------------------------------------------------------------
// Implicit-GEMM 3x3x3 SAME conv via WMMA, shape-templated so all index
// divisions strength-reduce to mul/shift. One wave = 16(M)x16(Cout) tile.
// ---------------------------------------------------------------------------
template<int CIN, int D, int COUT, int KP, int NTILES>
__global__ void conv_wmma_kernel(const float* __restrict__ in, const _Float16* __restrict__ wf,
                                 const float* __restrict__ bias, float* __restrict__ out) {
    constexpr int S = D * D * D;
    constexpr int K = CIN * 27;
    const int wave = (blockIdx.x * blockDim.x + threadIdx.x) >> 5;
    const int lane = threadIdx.x & 31;
    const int mt = wave / NTILES, nt = wave % NTILES;
    const int hi = lane >> 4;

    // A-side: this lane owns im2col row m (one output position)
    const int mA = mt * 16 + (lane & 15);
    const int n  = mA / S; const int s = mA - n * S;
    const int d0 = s / (D * D); const int r = s - d0 * (D * D);
    const int h0 = r / D; const int w0 = r - h0 * D;
    const float* inb = in + (size_t)n * CIN * S;
    const int kbaseA = hi ? 8 : 0;

    // B-side: this lane owns output-channel column
    const int co = nt * 16 + (lane & 15);
    const int kbaseB = hi ? 8 : 0;
    const _Float16* wrow = wf + (size_t)co * KP;

    float8 acc = {};
    for (int kc = 0; kc < KP; kc += 32) {
        __builtin_prefetch((const void*)(wrow + kc + 32), 0, 0);
        half16 a, bm;
#pragma unroll
        for (int i = 0; i < 16; ++i) {
            int k = kc + kbaseA + (i < 8 ? i : i + 8);
            float v = 0.0f;
            if (k < K) {
                int ci = k / 27; int rr = k - ci * 27;
                int kd = rr / 9; int r2 = rr - kd * 9;
                int kh = r2 / 3; int kw = r2 - kh * 3;
                int dd = d0 + kd - 1, hh = h0 + kh - 1, ww = w0 + kw - 1;
                if (((unsigned)dd < (unsigned)D) & ((unsigned)hh < (unsigned)D) &
                    ((unsigned)ww < (unsigned)D))
                    v = inb[ci * S + (dd * D + hh) * D + ww];
            }
            a[i] = (_Float16)v;
        }
#pragma unroll
        for (int i = 0; i < 16; ++i) bm[i] = wrow[kc + kbaseB + (i < 8 ? i : i + 8)];
        acc = __builtin_amdgcn_wmma_f32_16x16x32_f16(false, a, false, bm,
                                                     (short)0, acc, false, false);
    }
    const float bv = bias[co];
#pragma unroll
    for (int rr = 0; rr < 8; ++rr) {
        int m = mt * 16 + rr + (hi << 3);
        int nn = m / S; int ss = m - nn * S;
        out[((size_t)nn * COUT + co) * S + ss] = acc[rr] + bv;
    }
}

// ---------------------------------------------------------------------------
// Dense GEMM via WMMA: out[64][NV] = A[64][K] * Wf16[Np][KP]^T + bias.
// ---------------------------------------------------------------------------
template<int K, int KP, int NV, int NTILES, int LEAKY>
__global__ void dense_wmma_kernel(const float* __restrict__ A, const _Float16* __restrict__ wf,
                                  const float* __restrict__ bias, float* __restrict__ out) {
    const int wave = (blockIdx.x * blockDim.x + threadIdx.x) >> 5;
    const int lane = threadIdx.x & 31;
    const int mt = wave / NTILES, nt = wave % NTILES;
    const int hi = lane >> 4;

    const int mA = mt * 16 + (lane & 15);
    const float* arow = A + (size_t)mA * K;
    const int kbaseA = hi ? 8 : 0;

    const int co = nt * 16 + (lane & 15);
    const int kbaseB = hi ? 8 : 0;
    const _Float16* wrow = wf + (size_t)co * KP;

    float8 acc = {};
    for (int kc = 0; kc < KP; kc += 32) {
        __builtin_prefetch((const void*)(wrow + kc + 32), 0, 0);
        half16 a, bm;
#pragma unroll
        for (int i = 0; i < 16; ++i) {
            int k = kc + kbaseA + (i < 8 ? i : i + 8);
            a[i] = (_Float16)((k < K) ? arow[k] : 0.0f);
        }
#pragma unroll
        for (int i = 0; i < 16; ++i) bm[i] = wrow[kc + kbaseB + (i < 8 ? i : i + 8)];
        acc = __builtin_amdgcn_wmma_f32_16x16x32_f16(false, a, false, bm,
                                                     (short)0, acc, false, false);
    }
    if (co < NV) {
        const float bv = bias[co];
#pragma unroll
        for (int rr = 0; rr < 8; ++rr) {
            int m = mt * 16 + rr + (hi << 3);
            float v = acc[rr] + bv;
            if (LEAKY) v = (v >= 0.0f) ? v : 0.01f * v;
            out[(size_t)m * NV + co] = v;
        }
    }
}

// ---------------------------------------------------------------------------
// 2x2x2 max pool, stride 2, VALID.
// ---------------------------------------------------------------------------
__global__ void pool_kernel(const float* __restrict__ in, float* __restrict__ out,
                            int Din, int total) {
    int i = blockIdx.x * blockDim.x + threadIdx.x;
    if (i >= total) return;
    const int Do = Din >> 1;
    const int So = Do * Do * Do, Si = Din * Din * Din;
    int s = i % So; int nc = i / So;
    int d = s / (Do * Do); int r = s - d * Do * Do;
    int h = r / Do; int w = r - h * Do;
    const float* p = in + (size_t)nc * Si;
    float m = -INFINITY;
#pragma unroll
    for (int a = 0; a < 2; ++a)
#pragma unroll
        for (int b = 0; b < 2; ++b)
#pragma unroll
            for (int c = 0; c < 2; ++c) {
                float v = p[((2 * d + a) * Din + (2 * h + b)) * Din + (2 * w + c)];
                m = v > m ? v : m;
            }
    out[i] = m;
}

// ---------------------------------------------------------------------------
// BN stats over (N, spatial) per channel: one block per channel.
// ---------------------------------------------------------------------------
__global__ void bnstats_kernel(const float* __restrict__ x, float* __restrict__ mean,
                               float* __restrict__ rstd, int C, int S, int Nb) {
    const int c = blockIdx.x, tid = threadIdx.x;
    __shared__ float s1[256], s2[256];
    float a = 0.0f, b = 0.0f;
    const int tot = Nb * S;
    for (int i = tid; i < tot; i += 256) {
        int n = i / S; int s = i - n * S;
        float v = x[((size_t)n * C + c) * S + s];
        a += v; b += v * v;
    }
    s1[tid] = a; s2[tid] = b; __syncthreads();
    for (int st = 128; st > 0; st >>= 1) {
        if (tid < st) { s1[tid] += s1[tid + st]; s2[tid] += s2[tid + st]; }
        __syncthreads();
    }
    if (tid == 0) {
        float m = s1[0] / (float)tot;
        float var = s2[0] / (float)tot - m * m;
        mean[c] = m; rstd[c] = rsqrtf(var + 1e-5f);
    }
}

// BN stats per-feature over 64 rows (MLP, axes=(0,)).
__global__ void bnstats_rows_kernel(const float* __restrict__ x, float* __restrict__ mean,
                                    float* __restrict__ rstd, int F) {
    int f = blockIdx.x * blockDim.x + threadIdx.x;
    if (f >= F) return;
    float a = 0.0f, b = 0.0f;
    for (int r = 0; r < NB_BATCH; ++r) { float v = x[(size_t)r * F + f]; a += v; b += v * v; }
    float m = a / (float)NB_BATCH;
    float var = b / (float)NB_BATCH - m * m;
    mean[f] = m; rstd[f] = rsqrtf(var + 1e-5f);
}

// BN apply + leaky, in place. Channel index = (i / S) % C (use S=1 for MLP).
__global__ void bnapply_kernel(float* __restrict__ x, const float* __restrict__ mean,
                               const float* __restrict__ rstd, const float* __restrict__ g,
                               const float* __restrict__ bb, int C, int S, int total) {
    int i = blockIdx.x * blockDim.x + threadIdx.x;
    if (i >= total) return;
    int c = (i / S) % C;
    float v = (x[i] - mean[c]) * rstd[c] * g[c] + bb[c];
    x[i] = (v >= 0.0f) ? v : 0.01f * v;
}

// ---------------------------------------------------------------------------
// Launch
// ---------------------------------------------------------------------------
extern "C" void kernel_launch(void* const* d_in, const int* in_sizes, int n_in,
                              void* d_out, int out_size, void* d_ws, size_t ws_size,
                              hipStream_t stream) {
    (void)in_sizes; (void)n_in; (void)out_size; (void)ws_size;
    const float* x       = (const float*)d_in[0];
    const float* conv_w0 = (const float*)d_in[2];
    const float* conv_b0 = (const float*)d_in[3];
    const float* bn_g0   = (const float*)d_in[4];
    const float* bn_b0   = (const float*)d_in[5];
    const float* conv_w1 = (const float*)d_in[6];
    const float* conv_b1 = (const float*)d_in[7];
    const float* bn_g1   = (const float*)d_in[8];
    const float* bn_b1   = (const float*)d_in[9];
    const float* conv_w2 = (const float*)d_in[10];
    const float* conv_b2 = (const float*)d_in[11];
    const float* bn_g2   = (const float*)d_in[12];
    const float* bn_b2   = (const float*)d_in[13];
    const float* mlp_w0  = (const float*)d_in[14];
    const float* mlp_b0  = (const float*)d_in[15];
    const float* bn1_g   = (const float*)d_in[16];
    const float* bn1_b   = (const float*)d_in[17];
    const float* mlp_w1  = (const float*)d_in[18];
    const float* mlp_b1  = (const float*)d_in[19];
    const float* bn2_g   = (const float*)d_in[20];
    const float* bn2_b   = (const float*)d_in[21];
    const float* mlp_w2  = (const float*)d_in[22];
    const float* mlp_b2  = (const float*)d_in[23];
    const float* head_w  = (const float*)d_in[24];
    const float* head_b  = (const float*)d_in[25];
    float* out = (float*)d_out;
    float* ws  = (float*)d_ws;

    // workspace layout (float offsets)
    const size_t OFF_FIELDS = 0;          // 64*6*5832   = 2239488
    const size_t OFF_C0     = 2239488;    // 64*16*5832  = 5971968
    const size_t OFF_P0     = 8211456;    // 64*16*729   = 746496
    const size_t OFF_C1     = 0;          // reuse fields region: 64*32*729 = 1492992
    const size_t OFF_P1     = 8957952;    // 64*32*64    = 131072
    const size_t OFF_C2     = 9089024;    // 64*64*64    = 262144
    const size_t OFF_P2     = 9351168;    // 64*64*8     = 32768 (flatten h)
    const size_t OFF_H0     = 9383936;    // 64*512
    const size_t OFF_H1     = 9416704;    // 64*256
    const size_t OFF_H2     = 9433088;    // 64*100
    const size_t OFF_MEAN   = 9439488;    // 512
    const size_t OFF_RSTD   = 9440000;    // 512
    const size_t OFF_W16    = 9440512;    // f16 region begins (16B aligned)

    _Float16* wf = (_Float16*)(ws + OFF_W16);
    _Float16* wf_c0 = wf + 0;        // 16*192
    _Float16* wf_c1 = wf + 3072;     // 32*448
    _Float16* wf_c2 = wf + 17408;    // 64*864
    _Float16* wf_m0 = wf + 72704;    // 512*512
    _Float16* wf_m1 = wf + 334848;   // 256*512
    _Float16* wf_m2 = wf + 465920;   // 112*256
    _Float16* wf_hd = wf + 494592;   // 32*128

    float* mean = ws + OFF_MEAN;
    float* rstd = ws + OFF_RSTD;

    // weight prep (fp32 -> padded f16)
    prep_w_kernel<<<12,   256, 0, stream>>>(conv_w0, wf_c0, 16, 162, 16, 192);
    prep_w_kernel<<<56,   256, 0, stream>>>(conv_w1, wf_c1, 32, 432, 32, 448);
    prep_w_kernel<<<216,  256, 0, stream>>>(conv_w2, wf_c2, 64, 864, 64, 864);
    prep_w_kernel<<<1024, 256, 0, stream>>>(mlp_w0, wf_m0, 512, 512, 512, 512);
    prep_w_kernel<<<512,  256, 0, stream>>>(mlp_w1, wf_m1, 256, 512, 256, 512);
    prep_w_kernel<<<112,  256, 0, stream>>>(mlp_w2, wf_m2, 100, 256, 112, 256);
    prep_w_kernel<<<16,   256, 0, stream>>>(head_w, wf_hd, 20, 100, 32, 128);

    // blur -> fields (64,6,18,18,18)
    blur_kernel<<<64, 256, 0, stream>>>(x, ws + OFF_FIELDS);

    // conv0: (64,6,18^3) -> (64,16,18^3). M=373248 -> 23328 Mtiles, Ntiles=1.
    conv_wmma_kernel<6, 18, 16, 192, 1><<<5832, 128, 0, stream>>>(
        ws + OFF_FIELDS, wf_c0, conv_b0, ws + OFF_C0);
    pool_kernel<<<2916, 256, 0, stream>>>(ws + OFF_C0, ws + OFF_P0, 18, 746496);
    bnstats_kernel<<<16, 256, 0, stream>>>(ws + OFF_P0, mean, rstd, 16, 729, 64);
    bnapply_kernel<<<2916, 256, 0, stream>>>(ws + OFF_P0, mean, rstd, bn_g0, bn_b0,
                                             16, 729, 746496);

    // conv1: (64,16,9^3) -> (64,32,9^3). M=46656 -> 2916 Mtiles, Ntiles=2.
    conv_wmma_kernel<16, 9, 32, 448, 2><<<1458, 128, 0, stream>>>(
        ws + OFF_P0, wf_c1, conv_b1, ws + OFF_C1);
    pool_kernel<<<512, 256, 0, stream>>>(ws + OFF_C1, ws + OFF_P1, 9, 131072);
    bnstats_kernel<<<32, 256, 0, stream>>>(ws + OFF_P1, mean, rstd, 32, 64, 64);
    bnapply_kernel<<<512, 256, 0, stream>>>(ws + OFF_P1, mean, rstd, bn_g1, bn_b1,
                                            32, 64, 131072);

    // conv2: (64,32,4^3) -> (64,64,4^3). M=4096 -> 256 Mtiles, Ntiles=4.
    conv_wmma_kernel<32, 4, 64, 864, 4><<<256, 128, 0, stream>>>(
        ws + OFF_P1, wf_c2, conv_b2, ws + OFF_C2);
    pool_kernel<<<128, 256, 0, stream>>>(ws + OFF_C2, ws + OFF_P2, 4, 32768);
    bnstats_kernel<<<64, 256, 0, stream>>>(ws + OFF_P2, mean, rstd, 64, 8, 64);
    bnapply_kernel<<<128, 256, 0, stream>>>(ws + OFF_P2, mean, rstd, bn_g2, bn_b2,
                                            64, 8, 32768);

    // mlp0: [64,512]x[512,512] -> h0; bn1 + leaky
    dense_wmma_kernel<512, 512, 512, 32, 0><<<32, 128, 0, stream>>>(
        ws + OFF_P2, wf_m0, mlp_b0, ws + OFF_H0);
    bnstats_rows_kernel<<<2, 256, 0, stream>>>(ws + OFF_H0, mean, rstd, 512);
    bnapply_kernel<<<128, 256, 0, stream>>>(ws + OFF_H0, mean, rstd, bn1_g, bn1_b,
                                            512, 1, 32768);

    // mlp1: [64,512]x[256,512] -> h1; bn2 + leaky
    dense_wmma_kernel<512, 512, 256, 16, 0><<<16, 128, 0, stream>>>(
        ws + OFF_H0, wf_m1, mlp_b1, ws + OFF_H1);
    bnstats_rows_kernel<<<1, 256, 0, stream>>>(ws + OFF_H1, mean, rstd, 256);
    bnapply_kernel<<<64, 256, 0, stream>>>(ws + OFF_H1, mean, rstd, bn2_g, bn2_b,
                                           256, 1, 16384);

    // mlp2: [64,256]x[100,256] -> h2, fused leaky (no BN)
    dense_wmma_kernel<256, 256, 100, 7, 1><<<7, 128, 0, stream>>>(
        ws + OFF_H1, wf_m2, mlp_b2, ws + OFF_H2);

    // head: [64,100]x[20,100] -> out (64,20)
    dense_wmma_kernel<100, 128, 20, 2, 0><<<2, 128, 0, stream>>>(
        ws + OFF_H2, wf_hd, head_b, out);
}
